// MiningGNN_82755429860064
// MI455X (gfx1250) — matrix operand order repaired and checked by reading, not compiled, not run on hardware
//
#include <hip/hip_runtime.h>
#include <math.h>

typedef __attribute__((ext_vector_type(16))) _Float16 v16h;
typedef __attribute__((ext_vector_type(8)))  float    v8f;

#define HID 32
#define NEG_SLOPE 0.2f

// ---------- monotonic float<->uint encoding for atomicMax on floats ----------
static __device__ __forceinline__ unsigned f2ord(float f) {
    unsigned u = __float_as_uint(f);
    return (u & 0x80000000u) ? ~u : (u | 0x80000000u);
}
static __device__ __forceinline__ float ord2f(unsigned k) {
    return (k & 0x80000000u) ? __uint_as_float(k & 0x7fffffffu)
                             : __uint_as_float(~k);
}

// ---------- encoder: h = relu(x @ enc_W + enc_b), x is (n,5) ----------
__global__ void encoder_kernel(const float* __restrict__ x,
                               const float* __restrict__ W,   // 5 x 32
                               const float* __restrict__ b,
                               float* __restrict__ h, int n) {
    int idx = blockIdx.x * blockDim.x + threadIdx.x;
    if (idx >= n * HID) return;
    int node = idx >> 5, j = idx & 31;
    const float* xr = x + (size_t)node * 5;
    float acc = b[j];
#pragma unroll
    for (int k = 0; k < 5; ++k) acc = fmaf(xr[k], W[k * HID + j], acc);
    h[idx] = fmaxf(acc, 0.f);
}

// ---------- degree + edge_attr segment sums over dst ----------
__global__ void deg_loop_kernel(const int* __restrict__ dst,
                                const float* __restrict__ eattr,
                                float* __restrict__ deg,
                                float* __restrict__ lsum, int E) {
    int e = blockIdx.x * blockDim.x + threadIdx.x;
    if (e >= E) return;
    int d = dst[e];
    float4 a = *reinterpret_cast<const float4*>(eattr + (size_t)e * 4);
    atomicAdd(deg + d, 1.f);
    atomicAdd(lsum + (size_t)d * 4 + 0, a.x);
    atomicAdd(lsum + (size_t)d * 4 + 1, a.y);
    atomicAdd(lsum + (size_t)d * 4 + 2, a.z);
    atomicAdd(lsum + (size_t)d * 4 + 3, a.w);
}

__global__ void loop_div_kernel(float* __restrict__ lattr,
                                const float* __restrict__ deg, int n) {
    int i = blockIdx.x * blockDim.x + threadIdx.x;
    if (i >= n * 4) return;
    lattr[i] = lattr[i] / fmaxf(deg[i >> 2], 1.f);
}

// ---------- pack W (32x32 f32, row-major K x N) into WMMA B-operand layout ----------
// B 32x16 f16: lane L holds column (L&15); K = (L>>4)*8 + (j&7) + (j>>3)*16 for f16 element j.
// Two N-tiles (cols 0-15, cols 16-31) -> bpack[2][32][16] halfs.
__global__ void pack_W_kernel(const float* __restrict__ W,
                              _Float16* __restrict__ bpack) {
    int t = threadIdx.x;            // 0..63
    if (t >= 64) return;
    int ntile = t >> 5, lane = t & 31;
    int nc  = ntile * 16 + (lane & 15);
    int kb2 = (lane >> 4) * 8;
#pragma unroll
    for (int j = 0; j < 16; ++j)
        bpack[(size_t)t * 16 + j] = (_Float16)W[(kb2 + (j & 7) + ((j >> 3) << 4)) * HID + nc];
}

// ---------- wa = We @ att_e  (folds per-edge (attr@We)@att_e into dot4) ----------
__global__ void wa_kernel(const float* __restrict__ We,   // 4 x 32
                          const float* __restrict__ atte, // 32
                          float* __restrict__ wa) {
    int t = threadIdx.x;
    if (t >= 4) return;
    float acc = 0.f;
#pragma unroll
    for (int j = 0; j < HID; ++j) acc = fmaf(We[t * HID + j], atte[j], acc);
    wa[t] = acc;
}

// ---------- node transform via WMMA: xout = h @ W  (16 nodes / wave) ----------
__global__ void __launch_bounds__(256)
node_transform_wmma(const float* __restrict__ h,
                    const _Float16* __restrict__ bpack,  // [2][32][16]
                    float* __restrict__ xout, int n) {
    const int lane = threadIdx.x & 31;
    const int wave = threadIdx.x >> 5;
    const int tile = blockIdx.x * (blockDim.x >> 5) + wave;
    const int ntiles = (n + 15) >> 4;
    if (tile >= ntiles) return;          // wave-uniform: EXEC stays all-1s for WMMA
    const int base = tile << 4;
    const int kb   = (lane >> 4) << 3;   // K sub-block: 0 or 8
    int node = base + (lane & 15);
    if (node >= n) node = n - 1;         // clamp loads; stores guarded below
    const float* rowp = h + (size_t)node * HID;

    v16h a;
#pragma unroll
    for (int g = 0; g < 2; ++g) {        // K groups {kb..kb+7}, {16+kb..16+kb+7}
        float4 f0 = *reinterpret_cast<const float4*>(rowp + g * 16 + kb);
        float4 f1 = *reinterpret_cast<const float4*>(rowp + g * 16 + kb + 4);
        a[8 * g + 0] = (_Float16)f0.x; a[8 * g + 1] = (_Float16)f0.y;
        a[8 * g + 2] = (_Float16)f0.z; a[8 * g + 3] = (_Float16)f0.w;
        a[8 * g + 4] = (_Float16)f1.x; a[8 * g + 5] = (_Float16)f1.y;
        a[8 * g + 6] = (_Float16)f1.z; a[8 * g + 7] = (_Float16)f1.w;
    }
    v16h b0 = *reinterpret_cast<const v16h*>(bpack + (size_t)lane * 16);
    v16h b1 = *reinterpret_cast<const v16h*>(bpack + (size_t)(32 + lane) * 16);

    v8f c0 = {}; v8f c1 = {};
    c0 = __builtin_amdgcn_wmma_f32_16x16x32_f16(false, a, false, b0, (short)0, c0, false, false);
    c1 = __builtin_amdgcn_wmma_f32_16x16x32_f16(false, a, false, b1, (short)0, c1, false, false);

    // D layout: VGPR r -> row rbase+r, col (lane&15) [c0: cols 0-15, c1: cols 16-31]
    const int nc    = lane & 15;
    const int rbase = base + ((lane >> 4) << 3);
    float* outp = xout + (size_t)rbase * HID + nc;
    if (base + 16 <= n) {                // wave-uniform fast path: no per-row guards
#pragma unroll
        for (int r = 0; r < 8; ++r) {
            outp[r * HID]      = c0[r];
            outp[r * HID + 16] = c1[r];
        }
    } else {                             // only the final partial tile
#pragma unroll
        for (int r = 0; r < 8; ++r) {
            if (rbase + r < n) {
                outp[r * HID]      = c0[r];
                outp[r * HID + 16] = c1[r];
            }
        }
    }
}

// ---------- per-node attention scores a_src = x@att_src, a_dst = x@att_dst ----------
__global__ void attn_scores(const float* __restrict__ xo,
                            const float* __restrict__ atts,
                            const float* __restrict__ attd,
                            float* __restrict__ asrc,
                            float* __restrict__ adst, int n) {
    int i = blockIdx.x * blockDim.x + threadIdx.x;
    if (i >= n) return;
    const float4* r  = reinterpret_cast<const float4*>(xo + (size_t)i * HID);
    const float4* as = reinterpret_cast<const float4*>(atts);
    const float4* ad = reinterpret_cast<const float4*>(attd);
    float s = 0.f, d = 0.f;
#pragma unroll
    for (int q = 0; q < HID / 4; ++q) {
        float4 v = r[q], a = as[q], b = ad[q];
        s = fmaf(v.x, a.x, fmaf(v.y, a.y, fmaf(v.z, a.z, fmaf(v.w, a.w, s))));
        d = fmaf(v.x, b.x, fmaf(v.y, b.y, fmaf(v.z, b.z, fmaf(v.w, b.w, d))));
    }
    asrc[i] = s; adst[i] = d;
}

// ---------- pass 1: alpha = lrelu(asrc[s]+adst[d]+attr.wa), segment max ----------
__global__ void alpha_pass1(const int* __restrict__ src, const int* __restrict__ dst,
                            const float* __restrict__ eattr, const float* __restrict__ lattr,
                            const float* __restrict__ wa,
                            const float* __restrict__ asrc, const float* __restrict__ adst,
                            float* __restrict__ abuf, unsigned* __restrict__ amaxk,
                            int E, int n) {
    int e = blockIdx.x * blockDim.x + threadIdx.x;
    if (e >= E + n) return;
    int s, d; float4 ap;
    if (e < E) {
        s = src[e]; d = dst[e];
        ap = *reinterpret_cast<const float4*>(eattr + (size_t)e * 4);
    } else {
        s = e - E; d = s;
        ap = *reinterpret_cast<const float4*>(lattr + (size_t)(e - E) * 4);
    }
    float4 w = *reinterpret_cast<const float4*>(wa);
    float al = asrc[s] + adst[d] +
               fmaf(ap.x, w.x, fmaf(ap.y, w.y, fmaf(ap.z, w.z, ap.w * w.w)));
    al = (al > 0.f) ? al : NEG_SLOPE * al;
    abuf[e] = al;
    atomicMax(amaxk + d, f2ord(al));
}

// ---------- pass 2: ex = exp(alpha - amax[dst]); segment sum ----------
__global__ void alpha_pass2(const int* __restrict__ dst,
                            float* __restrict__ abuf,
                            const unsigned* __restrict__ amaxk,
                            float* __restrict__ denom, int E, int n) {
    int e = blockIdx.x * blockDim.x + threadIdx.x;
    if (e >= E + n) return;
    int d = (e < E) ? dst[e] : (e - E);
    float ex = expf(abuf[e] - ord2f(amaxk[d]));
    abuf[e] = ex;
    atomicAdd(denom + d, ex);
}

// ---------- pass 3: accum[dst] += (ex/denom[dst]) * x[src]  (32 lanes / edge) ----------
__global__ void scatter_pass3(const int* __restrict__ src, const int* __restrict__ dst,
                              const float* __restrict__ abuf, const float* __restrict__ denom,
                              const float* __restrict__ xo,
                              float* __restrict__ accum, int E, int n) {
    long long t = (long long)blockIdx.x * blockDim.x + threadIdx.x;
    if (t >= (long long)(E + n) * HID) return;
    int e = (int)(t >> 5), j = (int)(t & 31);
    int s, d;
    if (e < E) { s = src[e]; d = dst[e]; } else { s = e - E; d = s; }
    float coef = abuf[e] / denom[d];
    atomicAdd(accum + (size_t)d * HID + j, coef * xo[(size_t)s * HID + j]);
}

// ---------- finalize: h = (accum + bias), optional relu (float4 lanes) ----------
__global__ void finalize_kernel(const float* __restrict__ accum,
                                const float* __restrict__ bias,
                                float* __restrict__ hout, int n, int do_relu) {
    int idx = blockIdx.x * blockDim.x + threadIdx.x;   // over n*8 float4s
    if (idx >= n * (HID / 4)) return;
    float4 v = reinterpret_cast<const float4*>(accum)[idx];
    float4 b = reinterpret_cast<const float4*>(bias)[idx & 7];
    v.x += b.x; v.y += b.y; v.z += b.z; v.w += b.w;
    if (do_relu) {
        v.x = fmaxf(v.x, 0.f); v.y = fmaxf(v.y, 0.f);
        v.z = fmaxf(v.z, 0.f); v.w = fmaxf(v.w, 0.f);
    }
    reinterpret_cast<float4*>(hout)[idx] = v;
}

// ---------- decoder: log_softmax(h[:rows] @ dec_W + dec_b) ----------
__global__ void decoder_kernel(const float* __restrict__ h,
                               const float* __restrict__ W,  // 32 x 4
                               const float* __restrict__ b,
                               float* __restrict__ out, int rows) {
    int r = blockIdx.x * blockDim.x + threadIdx.x;
    if (r >= rows) return;
    const float4* hr = reinterpret_cast<const float4*>(h + (size_t)r * HID);
    float a0 = b[0], a1 = b[1], a2 = b[2], a3 = b[3];
#pragma unroll
    for (int q = 0; q < HID / 4; ++q) {
        float4 hv = hr[q];
        const float4* wr = reinterpret_cast<const float4*>(W + q * 16);
        float4 w0 = wr[0], w1 = wr[1], w2 = wr[2], w3 = wr[3];
        a0 = fmaf(hv.x, w0.x, fmaf(hv.y, w1.x, fmaf(hv.z, w2.x, fmaf(hv.w, w3.x, a0))));
        a1 = fmaf(hv.x, w0.y, fmaf(hv.y, w1.y, fmaf(hv.z, w2.y, fmaf(hv.w, w3.y, a1))));
        a2 = fmaf(hv.x, w0.z, fmaf(hv.y, w1.z, fmaf(hv.z, w2.z, fmaf(hv.w, w3.z, a2))));
        a3 = fmaf(hv.x, w0.w, fmaf(hv.y, w1.w, fmaf(hv.z, w2.w, fmaf(hv.w, w3.w, a3))));
    }
    float m = fmaxf(fmaxf(a0, a1), fmaxf(a2, a3));
    float s = expf(a0 - m) + expf(a1 - m) + expf(a2 - m) + expf(a3 - m);
    float ls = m + logf(s);
    float4 o;
    o.x = a0 - ls; o.y = a1 - ls; o.z = a2 - ls; o.w = a3 - ls;
    reinterpret_cast<float4*>(out)[r] = o;
}

static inline unsigned grid1(long long work, int block) {
    return (unsigned)((work + block - 1) / block);
}

extern "C" void kernel_launch(void* const* d_in, const int* in_sizes, int n_in,
                              void* d_out, int out_size, void* d_ws, size_t ws_size,
                              hipStream_t stream) {
    const float* x      = (const float*)d_in[0];
    const int*   ei     = (const int*)d_in[1];
    const float* eattr  = (const float*)d_in[2];
    const float* enc_W  = (const float*)d_in[4];
    const float* enc_b  = (const float*)d_in[5];
    const float* dec_W  = (const float*)d_in[18];
    const float* dec_b  = (const float*)d_in[19];

    const int n    = in_sizes[0] / 5;
    const int E    = in_sizes[1] / 2;
    const int Etot = E + n;
    const int rows = out_size / 4;
    const int* src = ei;
    const int* dst = ei + E;

    // ---- carve workspace (256B-aligned regions) ----
    char* p = (char*)d_ws;
    auto carve = [&](size_t bytes) -> void* {
        void* r = (void*)p;
        p += (bytes + 255) & ~(size_t)255;
        return r;
    };
    float*    hbuf  = (float*)carve((size_t)n * HID * 4);
    float*    xout  = (float*)carve((size_t)n * HID * 4);
    float*    accum = (float*)carve((size_t)n * HID * 4);
    float*    asrc  = (float*)carve((size_t)n * 4);
    float*    adst  = (float*)carve((size_t)n * 4);
    float*    deg   = (float*)carve((size_t)n * 4);
    float*    lattr = (float*)carve((size_t)n * 4 * 4);
    float*    abuf  = (float*)carve((size_t)Etot * 4);
    unsigned* amaxk = (unsigned*)carve((size_t)n * 4);
    float*    denom = (float*)carve((size_t)n * 4);
    float*    wa    = (float*)carve(64);
    _Float16* bpack = (_Float16*)carve(2 * 32 * 16 * sizeof(_Float16));
    (void)ws_size; (void)n_in;

    const int B = 256;

    // ---- encoder ----
    encoder_kernel<<<grid1((long long)n * HID, B), B, 0, stream>>>(x, enc_W, enc_b, hbuf, n);

    // ---- degree + mean edge_attr per dst (shared by both convs) ----
    hipMemsetAsync(deg,   0, (size_t)n * 4,     stream);
    hipMemsetAsync(lattr, 0, (size_t)n * 4 * 4, stream);
    deg_loop_kernel<<<grid1(E, B), B, 0, stream>>>(dst, eattr, deg, lattr, E);
    loop_div_kernel<<<grid1((long long)n * 4, B), B, 0, stream>>>(lattr, deg, n);

    // ---- one GAT conv ----
    auto run_conv = [&](const float* W, const float* atts, const float* attd,
                        const float* We, const float* atte, const float* bias,
                        int do_relu) {
        hipMemsetAsync(accum, 0, (size_t)n * HID * 4, stream);
        hipMemsetAsync(amaxk, 0, (size_t)n * 4,       stream);
        hipMemsetAsync(denom, 0, (size_t)n * 4,       stream);

        pack_W_kernel<<<1, 64, 0, stream>>>(W, bpack);
        wa_kernel<<<1, 32, 0, stream>>>(We, atte, wa);

        int ntiles = (n + 15) / 16;                     // 16 nodes per wave
        node_transform_wmma<<<grid1(ntiles, 8), 256, 0, stream>>>(hbuf, bpack, xout, n);

        attn_scores<<<grid1(n, B), B, 0, stream>>>(xout, atts, attd, asrc, adst, n);
        alpha_pass1<<<grid1(Etot, B), B, 0, stream>>>(src, dst, eattr, lattr, wa,
                                                      asrc, adst, abuf, amaxk, E, n);
        alpha_pass2<<<grid1(Etot, B), B, 0, stream>>>(dst, abuf, amaxk, denom, E, n);
        scatter_pass3<<<grid1((long long)Etot * HID, B), B, 0, stream>>>(
            src, dst, abuf, denom, xout, accum, E, n);
        finalize_kernel<<<grid1((long long)n * (HID / 4), B), B, 0, stream>>>(
            accum, bias, hbuf, n, do_relu);
    };

    run_conv((const float*)d_in[6],  (const float*)d_in[7],  (const float*)d_in[8],
             (const float*)d_in[9],  (const float*)d_in[10], (const float*)d_in[11], 1);
    run_conv((const float*)d_in[12], (const float*)d_in[13], (const float*)d_in[14],
             (const float*)d_in[15], (const float*)d_in[16], (const float*)d_in[17], 0);

    // ---- decoder + log_softmax ----
    decoder_kernel<<<grid1(rows, B), B, 0, stream>>>(hbuf, dec_W, dec_b, (float*)d_out, rows);
}
